// GIN_InfoMaxReg_52183852647109
// MI455X (gfx1250) — matrix operand order repaired and checked by reading, not compiled, not run on hardware
//
#include <hip/hip_runtime.h>

// ---------------- problem constants (match reference) ----------------
#define N_NODES   64000
#define N_EDGES   2048000
#define HID       256
#define NLAYER    5
#define N_PER_G   1000
#define NGRAPH    64           // N_NODES / N_PER_G
#define OUT       2
#define LH        (NLAYER * HID)   // 1280
#define BN_EPS    1e-5f
#define MH        ((size_t)N_NODES * HID)   // 16,384,000 elements

typedef __attribute__((ext_vector_type(8)))  float  v8f;
typedef __attribute__((ext_vector_type(8)))  __bf16 v8bf;
typedef __attribute__((ext_vector_type(16))) __bf16 v16bf;

// ---------------- small utility kernels ----------------
__global__ void zero_f32(float* __restrict__ p, int n) {
    int i = blockIdx.x * 256 + threadIdx.x;
    if (i < n) p[i] = 0.0f;
}

// pooled = (1 + eps[l]) * h   (the learn_eps residual term; atomics add on top)
__global__ void init_pooled(const float* __restrict__ h, const float* __restrict__ eps,
                            int l, float* __restrict__ pooled) {
    size_t i = (size_t)blockIdx.x * 256 + threadIdx.x;
    pooled[i] = (1.0f + eps[l]) * h[i];
}

// segment_sum: pooled[row[e]][:] += h[col[e]][:]  — edge x float4 parallel atomics.
// h (64 MB) is L2-resident on MI455X (192 MB L2), so gathers hit L2.
__global__ __launch_bounds__(256) void edge_scatter(const float* __restrict__ h,
                                                    const int* __restrict__ ei,
                                                    float* __restrict__ pooled) {
    size_t t = (size_t)blockIdx.x * 256 + threadIdx.x;  // N_EDGES*64 threads
    int e  = (int)(t >> 6);
    int f4 = (int)(t & 63) * 4;
    int r  = ei[e];
    int c  = ei[N_EDGES + e];
    const float4 v = *(const float4*)(h + (size_t)c * HID + f4);
    float* dst = pooled + (size_t)r * HID + f4;
    atomicAdd(dst + 0, v.x);
    atomicAdd(dst + 1, v.y);
    atomicAdd(dst + 2, v.z);
    atomicAdd(dst + 3, v.w);
}

__global__ void cvt_bf16(const float* __restrict__ src, __bf16* __restrict__ dst) {
    size_t i = (size_t)blockIdx.x * 256 + threadIdx.x;
    dst[i] = (__bf16)src[i];
}

// Bt[n][k] = (bf16) W[k][n]  — puts B fragments contiguous-in-K for WMMA lanes.
__global__ void prep_w(const float* __restrict__ W, __bf16* __restrict__ Bt) {
    int idx = blockIdx.x * 256 + threadIdx.x;   // 65536
    int k = idx >> 8, n = idx & 255;
    Bt[(size_t)n * HID + k] = (__bf16)W[(size_t)k * HID + n];
}

// ---------------- WMMA GEMM: C[M,256] = A[M,256](bf16) @ W(bf16, pre-transposed) + bias ----------------
// block = 256 thr = 8 waves; block tile 32(M)x256(N); wave tile 16(M)x64(N) = 4 accumulators.
__global__ __launch_bounds__(256) void gemm_bf16_wmma(const __bf16* __restrict__ A,
                                                      const __bf16* __restrict__ Bt,
                                                      const float* __restrict__ bias,
                                                      float* __restrict__ C) {
    const int wave   = threadIdx.x >> 5;
    const int lane   = threadIdx.x & 31;
    const int laneHi = lane >> 4;       // half-wave select
    const int lane16 = lane & 15;
    const int msub = wave >> 2;         // 0..1
    const int nsub = wave & 3;          // 0..3
    const int m0 = blockIdx.x * 32 + msub * 16;
    const int n0 = nsub * 64;

    v8f acc0 = {}, acc1 = {}, acc2 = {}, acc3 = {};

    const __bf16* aptr = A + (size_t)(m0 + lane16) * HID;

    for (int k0 = 0; k0 < HID; k0 += 32) {
        // A fragment (16x32 bf16): lane<16 -> K[k0..k0+7],[k0+16..k0+23]; lane>=16 -> +8
        const int kA = k0 + laneHi * 8;
        v8bf alo = *(const v8bf*)(aptr + kA);
        v8bf ahi = *(const v8bf*)(aptr + kA + 16);
        v16bf a;
#pragma unroll
        for (int i = 0; i < 8; ++i) { a[i] = alo[i]; a[8 + i] = ahi[i]; }

        // B fragments (32x16 bf16 each): lane holds column n, K [kB..kB+7],[kB+16..kB+23]
        const int kB = k0 + laneHi * 8;
#pragma unroll
        for (int j = 0; j < 4; ++j) {
            const __bf16* bptr = Bt + (size_t)(n0 + j * 16 + lane16) * HID + kB;
            v8bf b0 = *(const v8bf*)(bptr);
            v8bf b1 = *(const v8bf*)(bptr + 16);
            v16bf b;
#pragma unroll
            for (int i = 0; i < 8; ++i) { b[i] = b0[i]; b[8 + i] = b1[i]; }
            v8f& acc = (j == 0) ? acc0 : (j == 1) ? acc1 : (j == 2) ? acc2 : acc3;
            acc = __builtin_amdgcn_wmma_f32_16x16x32_bf16(
                false, a, false, b, (short)0, acc, false, false);
        }
    }

    // epilogue: C/D layout — VGPR r: lanes0-15 row=r, lanes16-31 row=r+8; col=lane16
#pragma unroll
    for (int j = 0; j < 4; ++j) {
        const v8f acc = (j == 0) ? acc0 : (j == 1) ? acc1 : (j == 2) ? acc2 : acc3;
        const int ncol = n0 + j * 16 + lane16;
        const float bv = bias[ncol];
#pragma unroll
        for (int r = 0; r < 8; ++r) {
            const int row = m0 + r + laneHi * 8;
            C[(size_t)row * HID + ncol] = acc[r] + bv;
        }
    }
}

// ---------------- batch-norm statistics: per-column sum & sum-of-squares ----------------
__global__ __launch_bounds__(256) void col_stats(const float* __restrict__ y,
                                                 float* __restrict__ sums) {
    const int col = threadIdx.x;
    const size_t r0 = (size_t)blockIdx.x * 250;     // 256 blocks * 250 rows = 64000
    float s = 0.0f, s2 = 0.0f;
    for (int i = 0; i < 250; ++i) {
        float v = y[(r0 + i) * HID + col];
        s += v; s2 += v * v;
    }
    atomicAdd(&sums[col], s);
    atomicAdd(&sums[HID + col], s2);
}

// BN (training mode, biased var) + ReLU; output either bf16 (feeds next GEMM) or f32
template <bool BF16OUT>
__global__ __launch_bounds__(256) void bn_relu(const float* __restrict__ y,
                                               const float* __restrict__ sums,
                                               const float* __restrict__ g,
                                               const float* __restrict__ b,
                                               void* __restrict__ outp) {
    size_t i = (size_t)blockIdx.x * 256 + threadIdx.x;
    const int col = threadIdx.x;   // 256 threads/block, stride-256 rows
    const float inv_n = 1.0f / (float)N_NODES;
    float mu  = sums[col] * inv_n;
    float var = sums[HID + col] * inv_n - mu * mu;
    float v = (y[i] - mu) * rsqrtf(var + BN_EPS) * g[col] + b[col];
    v = fmaxf(v, 0.0f);
    if (BF16OUT) ((__bf16*)outp)[i] = (__bf16)v;
    else         ((float*)outp)[i]  = v;
}

// ---------------- head ----------------
// ph[l][g][c] = mean over the graph's 1000 nodes
__global__ __launch_bounds__(256) void graph_pool(const float* __restrict__ hidden,
                                                  float* __restrict__ ph) {
    const int l = blockIdx.x / NGRAPH;
    const int g = blockIdx.x % NGRAPH;
    const int c = threadIdx.x;
    const float* base = hidden + (size_t)l * MH + (size_t)g * N_PER_G * HID + c;
    float s = 0.0f;
    for (int i = 0; i < N_PER_G; ++i) s += base[(size_t)i * HID];
    ph[((size_t)l * NGRAPH + g) * HID + c] = s * (1.0f / (float)N_PER_G);
}

// c_logit[g][o] = sum_l ph[l][g] @ pred_w[l] + pred_b[l]
__global__ void head_logit(const float* __restrict__ ph, const float* __restrict__ pw,
                           const float* __restrict__ pb, float* __restrict__ out) {
    int t = threadIdx.x;
    if (t >= NGRAPH * OUT) return;
    int g = t >> 1, o = t & 1;
    float acc = 0.0f;
    for (int l = 0; l < NLAYER; ++l) {
        acc += pb[l * OUT + o];
        const float* p = ph + ((size_t)l * NGRAPH + g) * HID;
        const float* w = pw + (size_t)l * HID * OUT + o;
        for (int k = 0; k < HID; ++k) acc += p[k] * w[k * OUT];
    }
    out[g * OUT + o] = acc;
}

// cbuf[g][l*256+c] = sigmoid(ph[l][g][c])  (summary vector, graph-major)
__global__ void sigmoid_c(const float* __restrict__ ph, float* __restrict__ cbuf) {
    int i = blockIdx.x * 256 + threadIdx.x;        // NLAYER*NGRAPH*HID = 81920
    int l = i / (NGRAPH * HID);
    int rem = i % (NGRAPH * HID);
    int g = rem >> 8, c = rem & 255;
    float v = ph[i];
    cbuf[(size_t)g * LH + l * HID + c] = 1.0f / (1.0f + __expf(-v));
}

// WcT[g][d] = sum_e disc_w[d][e] * c[g][e]   (exploits repeat-structure of c_x)
__global__ __launch_bounds__(256) void wc_gemm(const float* __restrict__ dw,
                                               const float* __restrict__ cbuf,
                                               float* __restrict__ WcT) {
    int i = blockIdx.x * 256 + threadIdx.x;        // NGRAPH*LH = 81920
    int g = i / LH, d = i % LH;
    const float* wr = dw + (size_t)d * LH;
    const float* cr = cbuf + (size_t)g * LH;
    float s = 0.0f;
    for (int e = 0; e < LH; ++e) s += wr[e] * cr[e];
    WcT[i] = s;
}

// sc1[n] = n_f[n] . WcT[g(n)] + b ;  sc2[n] = n_f[shuf[n]] . WcT[g(n)] + b
// one wave32 per node, lane-strided coalesced reads, shuffle reduction
__global__ __launch_bounds__(256) void disc_score(const float* __restrict__ hidden,
                                                  const float* __restrict__ WcT,
                                                  const int* __restrict__ shuf,
                                                  const float* __restrict__ disc_b,
                                                  float* __restrict__ out) {
    const int wave = threadIdx.x >> 5;
    const int lane = threadIdx.x & 31;
    const int n = blockIdx.x * 8 + wave;           // 8000 blocks * 8 waves = 64000
    const int g  = n / N_PER_G;
    const int n2 = shuf[n];                        // reference indexes n_f by graph ids (<64)
    const float* wrow = WcT + (size_t)g * LH;
    float a1 = 0.0f, a2 = 0.0f;
    for (int l = 0; l < NLAYER; ++l) {
        const float* h1 = hidden + (size_t)l * MH + (size_t)n  * HID;
        const float* h2 = hidden + (size_t)l * MH + (size_t)n2 * HID;
        const float* w  = wrow + l * HID;
        for (int k = lane; k < HID; k += 32) {
            float wv = w[k];
            a1 += h1[k] * wv;
            a2 += h2[k] * wv;
        }
    }
#pragma unroll
    for (int off = 16; off; off >>= 1) {
        a1 += __shfl_down(a1, off);
        a2 += __shfl_down(a2, off);
    }
    if (lane == 0) {
        out[n]           = a1 + disc_b[0];
        out[N_NODES + n] = a2 + disc_b[0];
    }
}

// ---------------- launcher ----------------
extern "C" void kernel_launch(void* const* d_in, const int* in_sizes, int n_in,
                              void* d_out, int out_size, void* d_ws, size_t ws_size,
                              hipStream_t stream) {
    const float* x        = (const float*)d_in[0];
    const int*   edge_idx = (const int*)  d_in[1];
    const int*   shuf     = (const int*)  d_in[2];
    const float* eps      = (const float*)d_in[3];
    const float* mlp_w1   = (const float*)d_in[4];
    const float* mlp_b1   = (const float*)d_in[5];
    const float* mlp_bn_g = (const float*)d_in[6];
    const float* mlp_bn_b = (const float*)d_in[7];
    const float* mlp_w2   = (const float*)d_in[8];
    const float* mlp_b2   = (const float*)d_in[9];
    const float* bn_g     = (const float*)d_in[10];
    const float* bn_b     = (const float*)d_in[11];
    const float* pred_w   = (const float*)d_in[12];
    const float* pred_b   = (const float*)d_in[13];
    const float* disc_w   = (const float*)d_in[14];
    const float* disc_b   = (const float*)d_in[15];
    float* out = (float*)d_out;

    // workspace carve-up
    float* hidden = (float*)d_ws;                    // NLAYER * MH f32
    float* pooled = hidden + (size_t)NLAYER * MH;    // MH f32
    float* ybuf   = pooled + MH;                     // MH f32
    float* ph     = ybuf + MH;                       // NLAYER*NGRAPH*HID
    float* cbuf   = ph   + (size_t)NLAYER * NGRAPH * HID;
    float* WcT    = cbuf + (size_t)NGRAPH * LH;
    float* sums   = WcT  + (size_t)NGRAPH * LH;      // 512 f32
    __bf16* abf   = (__bf16*)(sums + 512);           // MH bf16
    __bf16* tbf   = abf + MH;                        // MH bf16
    __bf16* w1bf  = tbf + MH;                        // 65536 bf16
    __bf16* w2bf  = w1bf + HID * HID;

    const int nElemBlocks = (int)(MH / 256);          // 64000
    const int edgeBlocks  = N_EDGES * 64 / 256;       // 512000
    const int gemmBlocks  = N_NODES / 32;             // 2000

    for (int l = 0; l < NLAYER; ++l) {
        const float* hin = (l == 0) ? x : (hidden + (size_t)(l - 1) * MH);
        float* hout = hidden + (size_t)l * MH;

        // aggregation: pooled = (1+eps)*h + segment_sum(h[col] -> row)
        init_pooled<<<nElemBlocks, 256, 0, stream>>>(hin, eps, l, pooled);
        edge_scatter<<<edgeBlocks, 256, 0, stream>>>(hin, edge_idx, pooled);
        cvt_bf16<<<nElemBlocks, 256, 0, stream>>>(pooled, abf);

        // MLP layer 1: Linear -> BN -> ReLU (bf16 out for next GEMM)
        prep_w<<<HID * HID / 256, 256, 0, stream>>>(mlp_w1 + (size_t)l * HID * HID, w1bf);
        gemm_bf16_wmma<<<gemmBlocks, 256, 0, stream>>>(abf, w1bf, mlp_b1 + (size_t)l * HID, ybuf);
        zero_f32<<<2, 256, 0, stream>>>(sums, 512);
        col_stats<<<256, 256, 0, stream>>>(ybuf, sums);
        bn_relu<true><<<nElemBlocks, 256, 0, stream>>>(ybuf, sums,
            mlp_bn_g + (size_t)l * HID, mlp_bn_b + (size_t)l * HID, tbf);

        // MLP layer 2: Linear -> outer BN -> ReLU (f32 out = hidden_rep[l])
        prep_w<<<HID * HID / 256, 256, 0, stream>>>(mlp_w2 + (size_t)l * HID * HID, w2bf);
        gemm_bf16_wmma<<<gemmBlocks, 256, 0, stream>>>(tbf, w2bf, mlp_b2 + (size_t)l * HID, ybuf);
        zero_f32<<<2, 256, 0, stream>>>(sums, 512);
        col_stats<<<256, 256, 0, stream>>>(ybuf, sums);
        bn_relu<false><<<nElemBlocks, 256, 0, stream>>>(ybuf, sums,
            bn_g + (size_t)l * HID, bn_b + (size_t)l * HID, hout);
    }

    // graph pooling + classifier head
    graph_pool<<<NLAYER * NGRAPH, 256, 0, stream>>>(hidden, ph);
    head_logit<<<1, 128, 0, stream>>>(ph, pred_w, pred_b, out);

    // DGI discriminator, restructured: s[n] = n_f[n] . (disc_w @ c[g(n)])
    sigmoid_c<<<NLAYER * NGRAPH * HID / 256, 256, 0, stream>>>(ph, cbuf);
    wc_gemm<<<NGRAPH * LH / 256, 256, 0, stream>>>(disc_w, cbuf, WcT);
    disc_score<<<N_NODES / 8, 256, 0, stream>>>(hidden, WcT, shuf, disc_b,
                                                out + NGRAPH * OUT);
}
